// MyLSTM_40355512713721
// MI455X (gfx1250) — matrix operand order repaired and checked by reading, not compiled, not run on hardware
//
#include <hip/hip_runtime.h>
#include <hip/hip_bf16.h>
#include <stdint.h>

// LSTM forward for B=64, T=256, ID=512, HD=1024 on gfx1250 (MI455X).
//
//  - One-time pack: W = [Wi; Wh] -> bf16, transposed to [N=4096][K=1536]
//    (12 MB, L2-resident for the whole recurrence); bc = bi + bh; c = 0.
//  - Per timestep (256 sequential launches): g = [x_t, h_{t-1}] @ W via
//    v_wmma_f32_16x16x32_bf16 (f32 accum). Each block owns 32 h-columns
//    => 128 g-columns (all 4 gates), so the F/I/A/O -> (c,h) pointwise is
//    block-local through LDS. h_t goes straight into Q_all (d_out).
//  - Double-buffered LDS K-chunks; weights staged with the CDNA5 async
//    global->LDS path (global_load_async_to_lds_b128 + s_wait_asynccnt),
//    activations staged through VGPRs (f32 -> bf16 conversion).
//  - All LDS fragments for a chunk are preloaded into VGPRs before the
//    WMMA burst so the 8 WMMAs issue back-to-back (one dscnt wait/chunk).

#define B_   64
#define T_   256
#define ID_  512
#define HD_  1024
#define KTOT 1536          // ID_ + HD_
#define NG   4096          // 4*HD_

typedef __attribute__((ext_vector_type(16))) __bf16 v16bf;
typedef __attribute__((ext_vector_type(8)))  __bf16 v8bf;
typedef __attribute__((ext_vector_type(8)))  float  v8f;

__device__ __forceinline__ float sigmoid_f(float x) {
    return 1.0f / (1.0f + __expf(-x));
}
__device__ __forceinline__ float tanh_f(float x) {
    return 1.0f - 2.0f / (__expf(2.0f * x) + 1.0f);
}
__device__ __forceinline__ uint32_t pk2(float a, float b) {
    unsigned short l = __builtin_bit_cast(unsigned short, (__bf16)a);
    unsigned short h = __builtin_bit_cast(unsigned short, (__bf16)b);
    return (uint32_t)l | ((uint32_t)h << 16);
}

// Generic->LDS offset: flat LDS aperture keeps the wave-relative LDS byte
// offset in addr[31:0] (ISA 10.2), so truncation gives the DS address.
__device__ __forceinline__ uint32_t lds_off(const void* p) {
    return (uint32_t)(uintptr_t)p;
}
__device__ __forceinline__ void async_load_b128(const void* gptr, const void* lptr) {
    asm volatile("global_load_async_to_lds_b128 %0, %1, off"
                 :: "v"(lds_off(lptr)), "v"((uint64_t)(uintptr_t)gptr)
                 : "memory");
}
__device__ __forceinline__ void wait_asynccnt0() {
    asm volatile("s_wait_asynccnt 0x0" ::: "memory");
}

// ---------------------------------------------------------------------------
// One-time weight repack: wc[n][k] = bf16( k<ID ? Wi[k][n] : Wh[k-ID][n] )
// ---------------------------------------------------------------------------
__global__ void lstm_pack_w(const float* __restrict__ Wi,
                            const float* __restrict__ Wh,
                            __bf16* __restrict__ wc) {
    int idx = blockIdx.x * blockDim.x + threadIdx.x;
    if (idx >= NG * KTOT) return;
    int n = idx / KTOT;
    int k = idx - n * KTOT;
    float v = (k < ID_) ? Wi[(size_t)k * NG + n]
                        : Wh[(size_t)(k - ID_) * NG + n];
    wc[idx] = (__bf16)v;
}

// bias combine + zero cell state (re-run every call: deterministic)
__global__ void lstm_init(const float* __restrict__ bi,
                          const float* __restrict__ bh,
                          float* __restrict__ bc,
                          float* __restrict__ c) {
    int idx = blockIdx.x * blockDim.x + threadIdx.x;
    if (idx < NG) bc[idx] = bi[idx] + bh[idx];
    if (idx < B_ * HD_) c[idx] = 0.0f;
}

// ---------------------------------------------------------------------------
// One recurrence step. Grid = HD_/32 = 32 blocks, 256 threads (8 wave32).
// ---------------------------------------------------------------------------
__launch_bounds__(256)
__global__ void lstm_step(const float* __restrict__ x,
                          const __bf16* __restrict__ wc,
                          const float* __restrict__ bc,
                          float* __restrict__ c,
                          float* __restrict__ q,       // d_out base (Q_all)
                          int t) {
    // Stage buffers (double-buffered) unioned with the gate buffer: the gate
    // buffer is only live after the K loop. 54 KB total.
    __shared__ union __attribute__((aligned(16))) {
        struct {
            __bf16 A[2][64][72];     // 2 x  9.0 KB
            __bf16 Bt[2][128][72];   // 2 x 18.0 KB
        } s;
        float G[64][132];            // 33.0 KB
    } sm;

    const int tid  = threadIdx.x;
    const int nb0  = blockIdx.x * 32;       // h-column base for this block
    const int lane = tid & 31;
    const int w    = tid >> 5;              // wave 0..7
    const int wm   = w & 3;                 // M tile
    const int wn   = w >> 2;                // N tile group (4 tiles)

    v8f acc[4];
    const v8f vzero = {0.f, 0.f, 0.f, 0.f, 0.f, 0.f, 0.f, 0.f};
#pragma unroll
    for (int qq = 0; qq < 4; ++qq) acc[qq] = vzero;

    const int klim = (t == 0) ? ID_ : KTOT; // h_{-1} == 0: skip Wh rows

    // --- staging thread mapping ---
    const int a_row = tid >> 2;             // 0..63  (batch row)
    const int a_k0  = (tid & 3) * 16;       // 16 k-values / thread
    const int b_col = tid >> 1;             // 0..127 (local g-column)
    const int b_k0  = (tid & 1) * 32;       // 32 k-values / thread

    const float* xrow = x + (size_t)(a_row * T_ + t) * ID_;
    const float* hrow = q + (size_t)(a_row * T_ + (t - 1)) * HD_; // iff t>0

    const int gate = b_col >> 5;
    const int jcol = b_col & 31;
    const __bf16* wcol = wc + (size_t)(gate * HD_ + nb0 + jcol) * KTOT;

    float4 fA[4];                            // in-flight A chunk (16 f32)

    auto loadA = [&](int kb) {               // global f32 -> regs
        const int kg = kb + a_k0;            // 16-aligned; never straddles ID_
        const float* src = (kg < ID_) ? (xrow + kg) : (hrow + (kg - ID_));
        fA[0] = ((const float4*)src)[0];
        fA[1] = ((const float4*)src)[1];
        fA[2] = ((const float4*)src)[2];
        fA[3] = ((const float4*)src)[3];
    };
    auto storeA = [&](int buf) {             // regs -> bf16 -> LDS
        uint4 p0, p1;
        p0.x = pk2(fA[0].x, fA[0].y); p0.y = pk2(fA[0].z, fA[0].w);
        p0.z = pk2(fA[1].x, fA[1].y); p0.w = pk2(fA[1].z, fA[1].w);
        p1.x = pk2(fA[2].x, fA[2].y); p1.y = pk2(fA[2].z, fA[2].w);
        p1.z = pk2(fA[3].x, fA[3].y); p1.w = pk2(fA[3].z, fA[3].w);
        *((uint4*)&sm.s.A[buf][a_row][a_k0])     = p0;
        *((uint4*)&sm.s.A[buf][a_row][a_k0 + 8]) = p1;
    };
    auto stageB = [&](int kb, int buf) {     // async global -> LDS, no VGPR data
        const __bf16* src = wcol + kb + b_k0;
#pragma unroll
        for (int i = 0; i < 4; ++i)
            async_load_b128(src + i * 8, &sm.s.Bt[buf][b_col][b_k0 + i * 8]);
    };

    // ---- prologue: stage chunk 0 into buffer 0
    loadA(0);
    stageB(0, 0);
    storeA(0);
    wait_asynccnt0();
    __syncthreads();

    const int mrow  = wm * 16 + (lane & 15);
    const int ahalf = (lane >> 4) * 8;      // A frag: K 0-7/16-23 vs 8-15/24-31
    const int bcf   = (lane & 15);
    const int bhalf = (lane >> 4) * 8;      // B frag: K 0-7/16-23 vs 8-15/24-31

    for (int kb = 0; kb < klim; kb += 64) {
        const int  cur  = (kb >> 6) & 1;
        const bool more = (kb + 64) < klim;

        // issue next chunk's loads before this chunk's math
        if (more) { loadA(kb + 64); stageB(kb + 64, cur ^ 1); }

        // ---- preload ALL fragments for this chunk (one dscnt wait), then
        //      fire the 8 WMMAs back-to-back into the matrix pipe.
        v16bf af[2];
        v16bf bfr[2][4];
#pragma unroll
        for (int s = 0; s < 2; ++s) {
            const int ks = s * 32;
            v8bf alo = *((const v8bf*)&sm.s.A[cur][mrow][ks + ahalf]);
            v8bf ahi = *((const v8bf*)&sm.s.A[cur][mrow][ks + ahalf + 16]);
            af[s] = __builtin_shufflevector(alo, ahi,
                0, 1, 2, 3, 4, 5, 6, 7, 8, 9, 10, 11, 12, 13, 14, 15);
#pragma unroll
            for (int qq = 0; qq < 4; ++qq) {
                const int nl = (wn * 4 + qq) * 16 + bcf;
                v8bf blo = *((const v8bf*)&sm.s.Bt[cur][nl][ks + bhalf]);
                v8bf bhi = *((const v8bf*)&sm.s.Bt[cur][nl][ks + bhalf + 16]);
                bfr[s][qq] = __builtin_shufflevector(blo, bhi,
                    0, 1, 2, 3, 4, 5, 6, 7, 8, 9, 10, 11, 12, 13, 14, 15);
            }
        }
#pragma unroll
        for (int s = 0; s < 2; ++s)
#pragma unroll
            for (int qq = 0; qq < 4; ++qq)
                acc[qq] = __builtin_amdgcn_wmma_f32_16x16x32_bf16(
                    false, af[s], false, bfr[s][qq], (short)0, acc[qq],
                    false, false);

        if (more) storeA(cur ^ 1);           // convert/park A for next chunk
        wait_asynccnt0();                    // B tile (next) landed in LDS
        __syncthreads();
    }

    // ---- spill accumulators to LDS gate buffer (C layout: VGPR r -> M=8*(lane/16)+r)
    {
        const int ncol = (lane & 15);
        const int mtop = wm * 16 + ((lane >> 4) ? 8 : 0);
#pragma unroll
        for (int qq = 0; qq < 4; ++qq) {
            const int nbase = (wn * 4 + qq) * 16 + ncol;
#pragma unroll
            for (int r = 0; r < 8; ++r)
                sm.G[mtop + r][nbase] = acc[qq][r];
        }
    }
    __syncthreads();

    // ---- pointwise gate update: 64 rows x 32 h-cols = 2048 elems
    const bool last = (t == T_ - 1);
    float* hT = q + (size_t)B_ * T_ * HD_;
    float* cT = hT + (size_t)B_ * HD_;
#pragma unroll
    for (int i = 0; i < 8; ++i) {
        int idx = tid + i * 256;
        int b = idx >> 5;
        int j = idx & 31;
        float gf = sm.G[b][j]      + bc[           nb0 + j];
        float gi = sm.G[b][32 + j] + bc[    HD_  + nb0 + j];
        float ga = sm.G[b][64 + j] + bc[2 * HD_  + nb0 + j];
        float go = sm.G[b][96 + j] + bc[3 * HD_  + nb0 + j];
        float f  = sigmoid_f(gf);
        float ii = sigmoid_f(gi);
        float aa = tanh_f(ga);
        float oo = sigmoid_f(go);
        int   ci = b * HD_ + nb0 + j;
        float cn = f * c[ci] + ii * aa;
        float hh = oo * tanh_f(cn);
        c[ci] = cn;
        q[(size_t)(b * T_ + t) * HD_ + nb0 + j] = hh;
        if (last) { hT[ci] = hh; cT[ci] = cn; }
    }
}

// ---------------------------------------------------------------------------
extern "C" void kernel_launch(void* const* d_in, const int* in_sizes, int n_in,
                              void* d_out, int out_size, void* d_ws, size_t ws_size,
                              hipStream_t stream) {
    (void)in_sizes; (void)n_in; (void)out_size; (void)ws_size;
    const float* x  = (const float*)d_in[0];
    const float* Wi = (const float*)d_in[1];
    const float* bi = (const float*)d_in[2];
    const float* Wh = (const float*)d_in[3];
    const float* bh = (const float*)d_in[4];
    float* out = (float*)d_out;

    // workspace layout: [ wc: 4096*1536 bf16 | bc: 4096 f32 | c: 65536 f32 ]
    __bf16* wc = (__bf16*)d_ws;
    const size_t wbytes = (size_t)NG * KTOT * sizeof(__bf16);   // 12,582,912 B
    float* bc = (float*)((char*)d_ws + wbytes);
    float* c  = bc + NG;

    lstm_pack_w<<<(NG * KTOT + 255) / 256, 256, 0, stream>>>(Wi, Wh, wc);
    lstm_init<<<(B_ * HD_ + 255) / 256, 256, 0, stream>>>(bi, bh, bc, c);

    for (int t = 0; t < T_; ++t)
        lstm_step<<<HD_ / 32, 256, 0, stream>>>(x, wc, bc, c, out, t);
}
